// LatentTherapyManifold_24756191494455
// MI455X (gfx1250) — matrix operand order repaired and checked by reading, not compiled, not run on hardware
//
#include <hip/hip_runtime.h>
#include <stdint.h>

// ---------------------------------------------------------------------------
// Types for gfx1250 WMMA (wave32). bf16 16x16x32 -> f32 accumulate.
// ---------------------------------------------------------------------------
typedef __attribute__((ext_vector_type(16))) __bf16 v16bf;
typedef __attribute__((ext_vector_type(8)))  __bf16 v8bf;
typedef __attribute__((ext_vector_type(8)))  float  v8f;

union U16bf { v16bf v; v8bf h[2]; };

static constexpr int ROWPAD = 520;   // 512 + 8 bf16 pad; row stride 1040B (16B aligned)

// ---------------------------------------------------------------------------
// Wave-level GEMM: D[16, NT*16] = A[16,512](LDS bf16) x W^T (global bf16) + bias.
// W stored [n][k] row-major with K = 512.  acc layout = WMMA C/D layout.
// Software-pipelined: B/A fragments for kt+1 are fetched into distinct
// registers while the WMMAs of kt execute (hides L2 latency).
// ---------------------------------------------------------------------------
template<int NT>
__device__ __forceinline__ void wave_gemm(const __bf16* A, const __bf16* W,
                                          const float* bias, int nt0, int lane,
                                          v8f* acc) {
  const int row = lane & 15;        // M row for A-frag, N col for B-frag & bias
  const int hi  = lane >> 4;

  const __bf16* ap = A + row * ROWPAD + hi * 8;
  const __bf16* wp[NT];
#pragma unroll
  for (int j = 0; j < NT; ++j)
    wp[j] = W + (size_t)((nt0 + j) * 16 + row) * 512 + hi * 8;

  // issue first fragment loads before touching bias (overlap with init)
  U16bf a_cur, b_cur[NT];
  a_cur.h[0] = *(const v8bf*)(ap);
  a_cur.h[1] = *(const v8bf*)(ap + 16);
#pragma unroll
  for (int j = 0; j < NT; ++j) {
    b_cur[j].h[0] = *(const v8bf*)(wp[j]);
    b_cur[j].h[1] = *(const v8bf*)(wp[j] + 16);
  }
#pragma unroll
  for (int j = 0; j < NT; ++j) {
    float b = bias[(nt0 + j) * 16 + row];
#pragma unroll
    for (int v = 0; v < 8; ++v) acc[j][v] = b;
  }

#pragma unroll
  for (int kt = 0; kt < 16; ++kt) {
    U16bf a_nxt, b_nxt[NT];
    if (kt < 15) {                       // compile-time after unroll
      const __bf16* ap2 = ap + (kt + 1) * 32;
      a_nxt.h[0] = *(const v8bf*)(ap2);
      a_nxt.h[1] = *(const v8bf*)(ap2 + 16);
#pragma unroll
      for (int j = 0; j < NT; ++j) {
        const __bf16* wp2 = wp[j] + (kt + 1) * 32;
        b_nxt[j].h[0] = *(const v8bf*)(wp2);
        b_nxt[j].h[1] = *(const v8bf*)(wp2 + 16);
      }
    }
#pragma unroll
    for (int j = 0; j < NT; ++j)
      acc[j] = __builtin_amdgcn_wmma_f32_16x16x32_bf16(
          false, a_cur.v, false, b_cur[j].v, (short)0, acc[j], false, false);
    if (kt < 15) {
      a_cur = a_nxt;
#pragma unroll
      for (int j = 0; j < NT; ++j) b_cur[j] = b_nxt[j];
    }
  }
}

template<int NT>
__device__ __forceinline__ void store_tiles(float* dst, const v8f* acc, int nt0, int lane) {
  const int nl = lane & 15, hi = lane >> 4;
#pragma unroll
  for (int j = 0; j < NT; ++j)
#pragma unroll
    for (int v = 0; v < 8; ++v)
      dst[(v + 8 * hi) * ROWPAD + (nt0 + j) * 16 + nl] = acc[j][v];
}

// LayerNorm over 512 cols of 16 rows in hs, activation, result -> xs (bf16).
// Caller must have barriered before call; this ends with a barrier.
__device__ __forceinline__ void ln_act(const float* hs, __bf16* xs,
                                       float* mu_s, float* rs_s,
                                       const float* g, const float* be,
                                       int tid, bool relu_mode) {
  const int r = tid >> 4, seg = tid & 15;
  const float* hrow = hs + r * ROWPAD + seg * 32;
  float s = 0.f, q = 0.f;
#pragma unroll
  for (int i = 0; i < 32; ++i) { float v = hrow[i]; s += v; q += v * v; }
#pragma unroll
  for (int m = 8; m >= 1; m >>= 1) {
    s += __shfl_xor(s, m, 32);
    q += __shfl_xor(q, m, 32);
  }
  if (seg == 0) {
    float mu  = s * (1.0f / 512.0f);
    float var = fmaxf(q * (1.0f / 512.0f) - mu * mu, 0.0f);
    mu_s[r] = mu;
    rs_s[r] = rsqrtf(var + 1e-5f);
  }
  __syncthreads();
  const float mu = mu_s[r], rstd = rs_s[r];
#pragma unroll 8
  for (int i = 0; i < 32; ++i) {
    int c = seg * 32 + i;
    float v = (hs[r * ROWPAD + c] - mu) * rstd * g[c] + be[c];
    float a = relu_mode ? fmaxf(v, 0.0f) : v * (1.0f / (1.0f + __expf(-v)));
    xs[r * ROWPAD + c] = (__bf16)a;
  }
  __syncthreads();
}

// ---------------------------------------------------------------------------
// Main persistent kernel: 4 workgroups x 256 threads, each owns 16 batch rows.
// ---------------------------------------------------------------------------
struct MainP {
  const __bf16 *WaT, *WbT, *W1T, *W2T, *noise;
  const float *cb, *sig;
  const float *g[4], *be[4], *bb[4];
  const float *dg, *dbe, *db1, *db2;
  float* out;
};

__global__ __launch_bounds__(256, 1) void k_main(MainP P) {
  __shared__ __bf16 xs[16 * ROWPAD];   // current x (bf16, GEMM A operand)
  __shared__ float  hs[16 * ROWPAD];   // fp32 pre-LN activations
  __shared__ float  mu_s[16], rs_s[16];

  const int tid  = threadIdx.x;
  const int lane = tid & 31;
  const int wv   = tid >> 5;           // 8 waves
  const int nl   = lane & 15, hi = lane >> 4;
  const int grow0 = blockIdx.x * 16;   // global batch row base

  // x0 = noise slot t=1000 (TEMPERATURE = 1)
  for (int i = tid; i < 16 * 512; i += 256) {
    int r = i >> 9, c = i & 511;
    xs[r * ROWPAD + c] = P.noise[((size_t)1000 << 15) + (size_t)(grow0 + r) * 512 + c];
  }
  __syncthreads();

  for (int t = 999; t >= 0; --t) {
    const float sg = P.sig[t];
    for (int blk = 0; blk < 4; ++blk) {
      // GEMM1: h = x @ WaT' + (te @ Wa_bot + ba)   (precomputed bias cb)
      const float* cbias = P.cb + ((size_t)t * 4 + blk) * 512;
      v8f acc[4];
      wave_gemm<4>(xs, P.WaT + (size_t)blk * 262144, cbias, wv * 4, lane, acc);
      store_tiles<4>(hs, acc, wv * 4, lane);
      __syncthreads();
      // LN + SiLU -> xs (overwrites old x; safe, GEMM1 reads done)
      ln_act(hs, xs, mu_s, rs_s, P.g[blk], P.be[blk], tid, false);
      // GEMM2: xnew = silu_h @ WbT' + bb
      v8f acc2[4];
      wave_gemm<4>(xs, P.WbT + (size_t)blk * 262144, P.bb[blk], wv * 4, lane, acc2);
      __syncthreads();   // all reads of xs finished before overwrite
      const bool addn = (blk == 3) && (t > 0);
#pragma unroll
      for (int j = 0; j < 4; ++j) {
        int n = (wv * 4 + j) * 16 + nl;
#pragma unroll
        for (int v = 0; v < 8; ++v) {
          int m = v + 8 * hi;
          float val = acc2[j][v];
          if (addn)
            val = fmaf(sg,
                       (float)P.noise[((size_t)t << 15) + (size_t)(grow0 + m) * 512 + n],
                       val);
          xs[m * ROWPAD + n] = (__bf16)val;
        }
      }
      __syncthreads();
    }
  }

  // Decoder: h = relu(LN(x @ dec_W1 + db1)); out = h @ dec_W2 + db2
  {
    v8f acc[4];
    wave_gemm<4>(xs, P.W1T, P.db1, wv * 4, lane, acc);
    store_tiles<4>(hs, acc, wv * 4, lane);
    __syncthreads();
    ln_act(hs, xs, mu_s, rs_s, P.dg, P.dbe, tid, true);
    v8f acc6[6];
    wave_gemm<6>(xs, P.W2T, P.db2, wv * 6, lane, acc6);
#pragma unroll
    for (int j = 0; j < 6; ++j) {
      int n = (wv * 6 + j) * 16 + nl;
#pragma unroll
      for (int v = 0; v < 8; ++v)
        P.out[(size_t)(grow0 + v + 8 * hi) * 768 + n] = acc6[j][v];
    }
  }
}

// ---------------------------------------------------------------------------
// Prep kernels
// ---------------------------------------------------------------------------
// dst[n*512 + k] = bf16(src[k*N + n])  (K fixed at 512)
__global__ void k_transp(const float* __restrict__ src, __bf16* __restrict__ dst,
                         int N, int total) {
  int idx = blockIdx.x * 256 + threadIdx.x;
  if (idx >= total) return;
  int n = idx / 512, k = idx - n * 512;
  dst[idx] = (__bf16)src[(size_t)k * N + n];
}

// thidden[t][k] = silu((t/1000)*t_W1[k] + t_b1[k])
__global__ void k_thid(const float* __restrict__ tW1, const float* __restrict__ tb1,
                       float* __restrict__ th) {
  int idx = blockIdx.x * 256 + threadIdx.x;      // < 1000*512
  int t = idx >> 9, k = idx & 511;
  float z = fmaf((float)t * 0.001f, tW1[k], tb1[k]);
  th[idx] = z * (1.0f / (1.0f + __expf(-z)));
}

// te[t][j] = sum_k thidden[t][k] * t_W2[k][j] + t_b2[j]
__global__ void k_te(const float* __restrict__ th, const float* __restrict__ tW2,
                     const float* __restrict__ tb2, float* __restrict__ te) {
  int idx = blockIdx.x * 256 + threadIdx.x;      // < 1000*512
  int t = idx >> 9, j = idx & 511;
  const float* h = th + (size_t)t * 512;
  float s = tb2[j];
  for (int k = 0; k < 512; ++k) s = fmaf(h[k], tW2[(size_t)k * 512 + j], s);
  te[idx] = s;
}

struct Ptr4 { const float* p[4]; };

// cb[t][blk][j] = sum_k te[t][k] * Wa[blk][512+k][j] + ba[blk][j]
__global__ void k_cbias(const float* __restrict__ te, Ptr4 Wa, Ptr4 ba,
                        float* __restrict__ cb) {
  int idx = blockIdx.x * 256 + threadIdx.x;      // < 1000*4*512
  int j = idx & 511, blk = (idx >> 9) & 3, t = idx >> 11;
  const float* W = Wa.p[blk];
  const float* ter = te + (size_t)t * 512;
  float s = ba.p[blk][j];
  for (int k = 0; k < 512; ++k) s = fmaf(ter[k], W[(size_t)(512 + k) * 512 + j], s);
  cb[idx] = s;
}

__global__ void k_sigma(float* sig) {
  int t = blockIdx.x * 256 + threadIdx.x;
  if (t > 1000) return;
  float b = 1e-4f + (float)t * ((0.02f - 1e-4f) / 999.0f);
  sig[t] = (t == 0) ? 0.0f : sqrtf(b);
}

// ---------------------------------------------------------------------------
// JAX threefry2x32 noise: noise[t] = normal(fold_in(key(42), t), (64,512))
// slot t=1000 doubles as x0.
// ---------------------------------------------------------------------------
__device__ __forceinline__ unsigned rotl32(unsigned x, int r) {
  return (x << r) | (x >> (32 - r));
}

__device__ __forceinline__ void threefry2x32(unsigned k0, unsigned k1,
                                             unsigned x0, unsigned x1,
                                             unsigned& o0, unsigned& o1) {
  unsigned k2 = k0 ^ k1 ^ 0x1BD11BDAu;
  x0 += k0; x1 += k1;
#define TF4(a,b,c,d) \
  x0 += x1; x1 = rotl32(x1,a); x1 ^= x0; \
  x0 += x1; x1 = rotl32(x1,b); x1 ^= x0; \
  x0 += x1; x1 = rotl32(x1,c); x1 ^= x0; \
  x0 += x1; x1 = rotl32(x1,d); x1 ^= x0;
  TF4(13,15,26,6)  x0 += k1; x1 += k2 + 1u;
  TF4(17,29,16,24) x0 += k2; x1 += k0 + 2u;
  TF4(13,15,26,6)  x0 += k0; x1 += k1 + 3u;
  TF4(17,29,16,24) x0 += k1; x1 += k2 + 4u;
  TF4(13,15,26,6)  x0 += k2; x1 += k0 + 5u;
#undef TF4
  o0 = x0; o1 = x1;
}

__device__ __forceinline__ float erfinv_approx(float x) {
  float w = -logf((1.0f - x) * (1.0f + x));
  float p;
  if (w < 5.0f) {
    w -= 2.5f;
    p = 2.81022636e-08f;
    p = fmaf(p, w, 3.43273939e-07f);  p = fmaf(p, w, -3.5233877e-06f);
    p = fmaf(p, w, -4.39150654e-06f); p = fmaf(p, w, 0.00021858087f);
    p = fmaf(p, w, -0.00125372503f);  p = fmaf(p, w, -0.00417768164f);
    p = fmaf(p, w, 0.246640727f);     p = fmaf(p, w, 1.50140941f);
  } else {
    w = sqrtf(w) - 3.0f;
    p = -0.000200214257f;
    p = fmaf(p, w, 0.000100950558f);  p = fmaf(p, w, 0.00134934322f);
    p = fmaf(p, w, -0.00367342844f);  p = fmaf(p, w, 0.00573950773f);
    p = fmaf(p, w, -0.0076224613f);   p = fmaf(p, w, 0.00943887047f);
    p = fmaf(p, w, 1.00167406f);      p = fmaf(p, w, 2.83297682f);
  }
  return p * x;
}

__device__ __forceinline__ float bits_to_normal(unsigned b) {
  float u01 = __uint_as_float((b >> 9) | 0x3f800000u) - 1.0f;  // [0,1)
  const float lo = -0.99999994f;
  float u = fmaxf(u01 * (1.0f - lo) + lo, lo);                  // (-1,1)
  return 1.41421356f * erfinv_approx(u);
}

__global__ void k_noise(__bf16* __restrict__ noise) {
  int i = blockIdx.x * 256 + threadIdx.x;   // 0..16383 (counter block id)
  int t = blockIdx.y + 1;                   // 1..1000
  unsigned fk0, fk1, o0, o1;
  threefry2x32(0u, 42u, 0u, (unsigned)t, fk0, fk1);           // fold_in(key, t)
  threefry2x32(fk0, fk1, (unsigned)i, (unsigned)(i + 16384), o0, o1);
  size_t base = (size_t)t << 15;
  noise[base + i]         = (__bf16)bits_to_normal(o0);
  noise[base + i + 16384] = (__bf16)bits_to_normal(o1);
}

// ---------------------------------------------------------------------------
// Host launch
// ---------------------------------------------------------------------------
extern "C" void kernel_launch(void* const* d_in, const int* in_sizes, int n_in,
                              void* d_out, int out_size, void* d_ws, size_t ws_size,
                              hipStream_t stream) {
  (void)n_in; (void)out_size; (void)ws_size;
  // --- resolve input ordering: dict-insertion vs jax-tree (sorted keys) ---
  int iWa[4], iba[4], ig[4], ibe[4], iWb[4], ibb[4];
  int i_tW1, i_tb1, i_tW2, i_tb2, i_dW1, i_db1, i_dg, i_dbe, i_dW2, i_db2;
  if (in_sizes[1] == 1024 * 512) {            // sorted: blocks first, each {Wa,Wb,ba,bb,beta,g}
    for (int b = 0; b < 4; ++b) {
      int s = 1 + 6 * b;
      iWa[b] = s; iWb[b] = s + 1; iba[b] = s + 2; ibb[b] = s + 3; ibe[b] = s + 4; ig[b] = s + 5;
    }
    i_dW1 = 25; i_dW2 = 26; i_db1 = 27; i_db2 = 28; i_dbe = 29; i_dg = 30;
    i_tW1 = 39; i_tW2 = 40; i_tb1 = 41; i_tb2 = 42;
  } else {                                    // insertion order
    i_tW1 = 9; i_tb1 = 10; i_tW2 = 11; i_tb2 = 12;
    for (int b = 0; b < 4; ++b) {
      int s = 13 + 6 * b;
      iWa[b] = s; iba[b] = s + 1; ig[b] = s + 2; ibe[b] = s + 3; iWb[b] = s + 4; ibb[b] = s + 5;
    }
    i_dW1 = 37; i_db1 = 38; i_dg = 39; i_dbe = 40; i_dW2 = 41; i_db2 = 42;
  }

  // --- workspace carve-up ---
  uint8_t* w = (uint8_t*)d_ws;
  size_t off = 0;
  auto carve = [&](size_t bytes) -> void* {
    off = (off + 255) & ~(size_t)255;
    void* r = w + off; off += bytes; return r;
  };
  __bf16* WaT = (__bf16*)carve((size_t)4 * 512 * 512 * 2);
  __bf16* WbT = (__bf16*)carve((size_t)4 * 512 * 512 * 2);
  __bf16* W1T = (__bf16*)carve((size_t)512 * 512 * 2);
  __bf16* W2T = (__bf16*)carve((size_t)768 * 512 * 2);
  float*  cbw = (float*)carve((size_t)1000 * 4 * 512 * 4);
  float*  thw = (float*)carve((size_t)1000 * 512 * 4);
  float*  tew = (float*)carve((size_t)1000 * 512 * 4);
  float*  sgw = (float*)carve((size_t)1004 * 4);
  __bf16* nzw = (__bf16*)carve((size_t)1001 * 32768 * 2);

  // --- weight transpose + fp32->bf16 ---
  for (int b = 0; b < 4; ++b) {
    k_transp<<<1024, 256, 0, stream>>>((const float*)d_in[iWa[b]], WaT + (size_t)b * 262144, 512, 262144);
    k_transp<<<1024, 256, 0, stream>>>((const float*)d_in[iWb[b]], WbT + (size_t)b * 262144, 512, 262144);
  }
  k_transp<<<1024, 256, 0, stream>>>((const float*)d_in[i_dW1], W1T, 512, 262144);
  k_transp<<<1536, 256, 0, stream>>>((const float*)d_in[i_dW2], W2T, 768, 393216);

  // --- time embeddings and per-step GEMM1 bias ---
  k_thid<<<2000, 256, 0, stream>>>((const float*)d_in[i_tW1], (const float*)d_in[i_tb1], thw);
  k_te<<<2000, 256, 0, stream>>>(thw, (const float*)d_in[i_tW2], (const float*)d_in[i_tb2], tew);
  Ptr4 WaP, baP;
  for (int b = 0; b < 4; ++b) { WaP.p[b] = (const float*)d_in[iWa[b]]; baP.p[b] = (const float*)d_in[iba[b]]; }
  k_cbias<<<8000, 256, 0, stream>>>(tew, WaP, baP, cbw);
  k_sigma<<<5, 256, 0, stream>>>(sgw);

  // --- all noise fields (t = 1..1000; slot 1000 = x0) ---
  k_noise<<<dim3(64, 1000), 256, 0, stream>>>(nzw);

  // --- main persistent diffusion kernel ---
  MainP P;
  P.WaT = WaT; P.WbT = WbT; P.W1T = W1T; P.W2T = W2T; P.noise = nzw;
  P.cb = cbw; P.sig = sgw;
  for (int b = 0; b < 4; ++b) {
    P.g[b]  = (const float*)d_in[ig[b]];
    P.be[b] = (const float*)d_in[ibe[b]];
    P.bb[b] = (const float*)d_in[ibb[b]];
  }
  P.dg  = (const float*)d_in[i_dg];
  P.dbe = (const float*)d_in[i_dbe];
  P.db1 = (const float*)d_in[i_db1];
  P.db2 = (const float*)d_in[i_db2];
  P.out = (float*)d_out;
  k_main<<<4, 256, 0, stream>>>(P);
}
